// MultiScaleCrossAttention_86912958202694
// MI455X (gfx1250) — compile-verified, matrix operand order checked by reading, not run on hardware
//
#include <hip/hip_runtime.h>

typedef __attribute__((ext_vector_type(16))) __bf16 v16bf;
typedef __attribute__((ext_vector_type(8)))  float  v8f;
typedef __attribute__((ext_vector_type(4)))  unsigned int v4u;
typedef unsigned short u16;
typedef unsigned int   u32;

#define NHEADS 12
#define HDIM   64
#define DIMX   768
#define BATCH  4
#define NTOK   3137              // 16*14*14 + 1
#define MROWS  (BATCH*NTOK)      // 12548
#define MPAD   12560             // ceil16(MROWS)
#define NQ     3137
#define NQPAD  3152
#define NKV    785               // 16*7*7 + 1
#define NKVPAD 800
#define QTILES 197               // NQPAD/16
#define SCALE_F 0.125f

__device__ __forceinline__ u16 f2bf(float f) {
  u32 u = __float_as_uint(f);
  u32 r = u + 0x7FFFu + ((u >> 16) & 1u);
  return (u16)(r >> 16);
}
__device__ __forceinline__ float bf2f(u16 h) {
  return __uint_as_float(((u32)h) << 16);
}

union FragU { u32 u[8]; v4u q[2]; v16bf bf; };

// Load a 16x32 bf16 fragment. Lane L -> row (L&15), k-group g=(L>>4).
// Per-lane element e holds k = (e&7) + (e>=8?16:0) + 8*g  (CDNA5 16-bit A/B layout).
// base must be 16B aligned with stride*2 % 16 == 0 (true for all call sites) so
// each fragment is exactly two global_load_b128 per lane.
__device__ __forceinline__ void load_frag(const u16* base, int stride, int lane, FragU& f) {
  int r = lane & 15, g = lane >> 4;
  const v4u* p = (const v4u*)(base + (size_t)r * stride);
  f.q[0] = p[g];       // dwords g*4 .. g*4+3
  f.q[1] = p[g + 2];   // dwords 8+g*4 .. 8+g*4+3
}

__device__ __forceinline__ v8f wmma_bf16(const FragU& a, const FragU& b, v8f c) {
  return __builtin_amdgcn_wmma_f32_16x16x32_bf16(false, a.bf, false, b.bf, (short)0, c, false, false);
}

// ---------------- conversion kernels ----------------
__global__ void convert_pad_kernel(const float* __restrict__ in, u16* __restrict__ out,
                                   int validRows, long long total) {
  long long i = (long long)blockIdx.x * blockDim.x + threadIdx.x;
  if (i >= total) return;
  long long row = i / DIMX;
  out[i] = (row < validRows) ? f2bf(in[i]) : (u16)0;
}

__global__ void convert_flat_kernel(const float* __restrict__ in, u16* __restrict__ out,
                                    long long n) {
  long long i = (long long)blockIdx.x * blockDim.x + threadIdx.x;
  if (i < n) out[i] = f2bf(in[i]);
}

__global__ void write_shape_kernel(float* __restrict__ out) {
  int i = threadIdx.x;
  if (i < 3) out[i] = (i == 0) ? 16.f : 14.f;
}

// ---------------- QKV GEMM: [MPAD,768] x [768,2304] -> [3][B][NH][N][64] f32 ----
__global__ void gemm_qkv_kernel(const u16* __restrict__ A, const u16* __restrict__ W,
                                float* __restrict__ out) {
  int wid  = (int)((blockIdx.x * blockDim.x + threadIdx.x) >> 5);
  int lane = threadIdx.x & 31;
  const int NT = (3 * DIMX) / 16; // 144 col tiles
  int tm = wid / NT, tn = wid % NT;
  if (tm >= MPAD / 16) return;
  const u16* arow = A + (size_t)(tm * 16) * DIMX;
  const u16* brow = W + (size_t)(tn * 16) * DIMX;   // W is [n,k] = qkv_w row-major
  v8f acc = {};
  for (int k0 = 0; k0 < DIMX; k0 += 32) {
    FragU a, b;
    __builtin_prefetch(brow + k0 + 32, 0, 3);
    load_frag(arow + k0, DIMX, lane, a);
    load_frag(brow + k0, DIMX, lane, b);
    acc = wmma_bf16(a, b, acc);
  }
  int g  = lane >> 4;
  int nn = tn * 16 + (lane & 15);
  int qi = nn / DIMX, rem = nn % DIMX;
  int h  = rem / HDIM, d = rem % HDIM;
  size_t slice = (size_t)BATCH * NHEADS * NTOK * HDIM;
  for (int j = 0; j < 8; ++j) {
    int m = tm * 16 + j + 8 * g;
    if (m < MROWS) {
      int b_ = m / NTOK, n = m % NTOK;
      out[(size_t)qi * slice + (((size_t)(b_ * NHEADS + h)) * NTOK + n) * HDIM + d] = acc[j];
    }
  }
}

// ---------------- depthwise 3x3 conv pool + LayerNorm(C=64) -> bf16 ----------
// in:  [B*NH][NTOK][64] f32 slice.  out: row-major [bh][NtokPad][64]  or
// transposed [bh][64][NtokPad] (for V so attention B-fragments are contiguous).
__global__ void pool_ln_kernel(const float* __restrict__ in, const float* __restrict__ cw,
                               const float* __restrict__ gamma, const float* __restrict__ beta,
                               u16* __restrict__ outp, int stride, int Hp, int Wp,
                               int Ntok, int NtokPad, int transposed) {
  int c  = threadIdx.x;             // channel 0..63
  int t  = blockIdx.x % NtokPad;    // output token
  int bh = blockIdx.x / NtokPad;    // b*NH + h
  size_t oidx = transposed ? (((size_t)bh * HDIM + c) * NtokPad + t)
                           : (((size_t)bh * NtokPad + t) * HDIM + c);
  if (t >= Ntok) { outp[oidx] = (u16)0; return; }   // block-uniform branch
  const float* ib = in + (size_t)bh * NTOK * HDIM;
  float val;
  if (t == 0) {
    val = ib[c];                                    // cls token passes through conv
  } else {
    int l  = t - 1;
    int tt = l / (Hp * Wp);
    int r  = l % (Hp * Wp);
    int hh = r / Wp, ww = r % Wp;
    float s = 0.f;
    for (int dy = 0; dy < 3; ++dy)
      for (int dx = 0; dx < 3; ++dx) {
        int yi = hh * stride + dy - 1, xi = ww * stride + dx - 1;
        if (yi >= 0 && yi < 14 && xi >= 0 && xi < 14) {
          int tin = 1 + (tt * 14 + yi) * 14 + xi;
          s += ib[(size_t)tin * HDIM + c] * cw[c * 9 + dy * 3 + dx];
        }
      }
    val = s;
  }
  __shared__ float r1[64], r2[64];
  r1[c] = val; r2[c] = val * val;
  __syncthreads();
  for (int off = 32; off; off >>= 1) {
    if (c < off) { r1[c] += r1[c + off]; r2[c] += r2[c + off]; }
    __syncthreads();
  }
  float mu  = r1[0] * (1.f / 64.f);
  float var = r2[0] * (1.f / 64.f) - mu * mu;
  float o   = (val - mu) * rsqrtf(var + 1e-5f) * gamma[c] + beta[c];
  outp[oidx] = f2bf(o);
}

// ---------------- flash attention: 1 wave per (b,h,qtile of 16) ----------------
__global__ void attn_kernel(const u16* __restrict__ qb,  // [B*NH][NQPAD][64]
                            const u16* __restrict__ kb,  // [B*NH][NKVPAD][64]
                            const u16* __restrict__ vt,  // [B*NH][64][NKVPAD]
                            u16* __restrict__ ob) {      // [MPAD][768]
  __shared__ __align__(16) u16 pbuf[4][16 * 32];
  int wslot = threadIdx.x >> 5;
  int lane  = threadIdx.x & 31;
  int w  = blockIdx.x * 4 + wslot;
  int qt = w % QTILES;
  int bh = w / QTILES;                       // 0..47
  const u16* qbh = qb + (size_t)bh * NQPAD * HDIM;
  const u16* kbh = kb + (size_t)bh * NKVPAD * HDIM;
  const u16* vbh = vt + (size_t)bh * HDIM * NKVPAD;

  FragU qf0, qf1;
  load_frag(qbh + (size_t)(qt * 16) * HDIM + 0,  HDIM, lane, qf0);
  load_frag(qbh + (size_t)(qt * 16) * HDIM + 32, HDIM, lane, qf1);

  float mi[8], li[8];
  v8f O0 = {}, O1 = {}, O2 = {}, O3 = {};
  for (int j = 0; j < 8; ++j) { mi[j] = -1e30f; li[j] = 0.f; }

  int g = lane >> 4, ln = lane & 15;
  u16* pb = pbuf[wslot];

  for (int c = 0; c < NKVPAD / 32; ++c) {
    v8f S0 = {}, S1 = {};
    FragU kf;
    load_frag(kbh + (size_t)(c * 32 +  0) * HDIM + 0,  HDIM, lane, kf); S0 = wmma_bf16(qf0, kf, S0);
    load_frag(kbh + (size_t)(c * 32 +  0) * HDIM + 32, HDIM, lane, kf); S0 = wmma_bf16(qf1, kf, S0);
    load_frag(kbh + (size_t)(c * 32 + 16) * HDIM + 0,  HDIM, lane, kf); S1 = wmma_bf16(qf0, kf, S1);
    load_frag(kbh + (size_t)(c * 32 + 16) * HDIM + 32, HDIM, lane, kf); S1 = wmma_bf16(qf1, kf, S1);

    int col0 = c * 32 + ln, col1 = col0 + 16;
    float s0[8], s1[8], p0[8], p1[8];
    for (int j = 0; j < 8; ++j) {
      s0[j] = (col0 < NKV) ? S0[j] * SCALE_F : -1e30f;
      s1[j] = (col1 < NKV) ? S1[j] * SCALE_F : -1e30f;
    }
    for (int j = 0; j < 8; ++j) {
      float t = fmaxf(s0[j], s1[j]);
      for (int msk = 8; msk; msk >>= 1) t = fmaxf(t, __shfl_xor(t, msk, 32));
      float nm = fmaxf(mi[j], t);
      float alpha = __expf(mi[j] - nm);
      p0[j] = __expf(s0[j] - nm);
      p1[j] = __expf(s1[j] - nm);
      float rs = p0[j] + p1[j];
      for (int msk = 8; msk; msk >>= 1) rs += __shfl_xor(rs, msk, 32);
      li[j] = li[j] * alpha + rs;
      mi[j] = nm;
      O0[j] *= alpha; O1[j] *= alpha; O2[j] *= alpha; O3[j] *= alpha;
    }
    // repack P (C-layout) -> A-fragment layout via LDS
    __syncthreads();
    for (int j = 0; j < 8; ++j) {
      int mrow = j + 8 * g;
      { int n = ln;      int la = mrow + 16 * ((n >> 3) & 1); int e = (n & 7) + ((n & 16) ? 8 : 0); pb[la * 16 + e] = f2bf(p0[j]); }
      { int n = ln + 16; int la = mrow + 16 * ((n >> 3) & 1); int e = (n & 7) + ((n & 16) ? 8 : 0); pb[la * 16 + e] = f2bf(p1[j]); }
    }
    __syncthreads();
    FragU pf;
    { const v4u* pp = (const v4u*)(pb + lane * 16);   // 32B aligned -> 2x ds_load_b128
      pf.q[0] = pp[0]; pf.q[1] = pp[1]; }
    FragU vf;
    load_frag(vbh + (size_t)(0 * 16) * NKVPAD + c * 32, NKVPAD, lane, vf); O0 = wmma_bf16(pf, vf, O0);
    load_frag(vbh + (size_t)(1 * 16) * NKVPAD + c * 32, NKVPAD, lane, vf); O1 = wmma_bf16(pf, vf, O1);
    load_frag(vbh + (size_t)(2 * 16) * NKVPAD + c * 32, NKVPAD, lane, vf); O2 = wmma_bf16(pf, vf, O2);
    load_frag(vbh + (size_t)(3 * 16) * NKVPAD + c * 32, NKVPAD, lane, vf); O3 = wmma_bf16(pf, vf, O3);
  }

  int b_ = bh / NHEADS, h = bh % NHEADS;
  for (int j = 0; j < 8; ++j) {
    int qi = qt * 16 + j + 8 * g;
    if (qi >= NQ) continue;
    float inv = 1.0f / li[j];
    size_t orow = ((size_t)(b_ * NQ + qi)) * DIMX + h * HDIM;
    float vals[4] = { O0[j] * inv, O1[j] * inv, O2[j] * inv, O3[j] * inv };
    for (int f = 0; f < 4; ++f) {
      int d = f * 16 + ln;
      float o = vals[f];
      if (qi >= 1) o += bf2f(qbh[(size_t)qi * HDIM + d]);   // residual: pooled+LN q
      ob[orow + d] = f2bf(o);
    }
  }
}

// ---------------- output projection: [MPAD,768] x [768,768] + bias -> f32 -------
__global__ void gemm_proj_kernel(const u16* __restrict__ A, const u16* __restrict__ W,
                                 const float* __restrict__ bias, float* __restrict__ out) {
  int wid  = (int)((blockIdx.x * blockDim.x + threadIdx.x) >> 5);
  int lane = threadIdx.x & 31;
  const int NT = DIMX / 16; // 48
  int tm = wid / NT, tn = wid % NT;
  if (tm >= MPAD / 16) return;
  const u16* arow = A + (size_t)(tm * 16) * DIMX;
  const u16* brow = W + (size_t)(tn * 16) * DIMX;   // proj_w is [n,k]
  v8f acc = {};
  for (int k0 = 0; k0 < DIMX; k0 += 32) {
    FragU a, b;
    __builtin_prefetch(brow + k0 + 32, 0, 3);
    load_frag(arow + k0, DIMX, lane, a);
    load_frag(brow + k0, DIMX, lane, b);
    acc = wmma_bf16(a, b, acc);
  }
  int g = lane >> 4, nn = tn * 16 + (lane & 15);
  float bv = bias[nn];
  for (int j = 0; j < 8; ++j) {
    int m = tm * 16 + j + 8 * g;
    if (m < MROWS) out[(size_t)m * DIMX + nn] = acc[j] + bv;
  }
}

extern "C" void kernel_launch(void* const* d_in, const int* in_sizes, int n_in,
                              void* d_out, int out_size, void* d_ws, size_t ws_size,
                              hipStream_t stream) {
  (void)in_sizes; (void)n_in; (void)out_size; (void)ws_size;
  const float* x      = (const float*)d_in[0];
  const float* y      = (const float*)d_in[1];
  const float* qkv_w  = (const float*)d_in[5];
  const float* proj_w = (const float*)d_in[6];
  const float* proj_b = (const float*)d_in[7];
  const float* pqw    = (const float*)d_in[8];
  const float* pkw    = (const float*)d_in[9];
  const float* pvw    = (const float*)d_in[10];
  const float* gq = (const float*)d_in[11]; const float* bq = (const float*)d_in[12];
  const float* gk = (const float*)d_in[13]; const float* bk = (const float*)d_in[14];
  const float* gv = (const float*)d_in[15]; const float* bv = (const float*)d_in[16];

  char* ws = (char*)d_ws;
  size_t off = 0;
  auto alloc = [&](size_t bytes) -> void* {
    void* p = ws + off;
    off = (off + bytes + 255) & ~(size_t)255;
    return p;
  };
  const size_t SL = (size_t)BATCH * NHEADS * NTOK * HDIM;   // floats per qkv slice

  u16*   xb   = (u16*)  alloc((size_t)MPAD * DIMX * 2);
  u16*   yb   = (u16*)  alloc((size_t)MPAD * DIMX * 2);
  u16*   wb   = (u16*)  alloc((size_t)3 * DIMX * DIMX * 2);
  u16*   pwb  = (u16*)  alloc((size_t)DIMX * DIMX * 2);
  float* qkvx = (float*)alloc(3 * SL * 4);
  float* qkvy = (float*)alloc(3 * SL * 4);
  u16*   qbf0 = (u16*)  alloc((size_t)48 * NQPAD * HDIM * 2);
  u16*   qbf1 = (u16*)  alloc((size_t)48 * NQPAD * HDIM * 2);
  u16*   kbf0 = (u16*)  alloc((size_t)48 * NKVPAD * HDIM * 2);
  u16*   kbf1 = (u16*)  alloc((size_t)48 * NKVPAD * HDIM * 2);
  u16*   vt0  = (u16*)  alloc((size_t)48 * HDIM * NKVPAD * 2);
  u16*   vt1  = (u16*)  alloc((size_t)48 * HDIM * NKVPAD * 2);
  // o buffers alias the (dead-by-then) qkvx fp32 region: pools finish first.
  u16* obf0 = (u16*)qkvx;
  u16* obf1 = (u16*)qkvx + (size_t)MPAD * DIMX;

  // 1) bf16 conversion
  {
    long long tot = (long long)MPAD * DIMX;
    int blks = (int)((tot + 255) / 256);
    convert_pad_kernel<<<blks, 256, 0, stream>>>(x, xb, MROWS, tot);
    convert_pad_kernel<<<blks, 256, 0, stream>>>(y, yb, MROWS, tot);
  }
  {
    long long n = (long long)3 * DIMX * DIMX;
    convert_flat_kernel<<<(int)((n + 255) / 256), 256, 0, stream>>>(qkv_w, wb, n);
  }
  {
    long long n = (long long)DIMX * DIMX;
    convert_flat_kernel<<<(int)((n + 255) / 256), 256, 0, stream>>>(proj_w, pwb, n);
  }

  // 2) QKV GEMMs (785 x 144 tiles, 8 waves/block)
  {
    int waves = (MPAD / 16) * ((3 * DIMX) / 16);
    int blks = waves / 8;
    gemm_qkv_kernel<<<blks, 256, 0, stream>>>(xb, wb, qkvx);
    gemm_qkv_kernel<<<blks, 256, 0, stream>>>(yb, wb, qkvy);
  }

  // 3) conv pool + LN (cross wiring: stream0 q from y, k/v from x; stream1 mirrored)
  pool_ln_kernel<<<48 * NQPAD,  64, 0, stream>>>(qkvy + 0 * SL, pqw, gq, bq, qbf0, 1, 14, 14, NQ,  NQPAD,  0);
  pool_ln_kernel<<<48 * NKVPAD, 64, 0, stream>>>(qkvx + 1 * SL, pkw, gk, bk, kbf0, 2,  7,  7, NKV, NKVPAD, 0);
  pool_ln_kernel<<<48 * NKVPAD, 64, 0, stream>>>(qkvx + 2 * SL, pvw, gv, bv, vt0,  2,  7,  7, NKV, NKVPAD, 1);
  pool_ln_kernel<<<48 * NQPAD,  64, 0, stream>>>(qkvx + 0 * SL, pqw, gq, bq, qbf1, 1, 14, 14, NQ,  NQPAD,  0);
  pool_ln_kernel<<<48 * NKVPAD, 64, 0, stream>>>(qkvy + 1 * SL, pkw, gk, bk, kbf1, 2,  7,  7, NKV, NKVPAD, 0);
  pool_ln_kernel<<<48 * NKVPAD, 64, 0, stream>>>(qkvy + 2 * SL, pvw, gv, bv, vt1,  2,  7,  7, NKV, NKVPAD, 1);

  // 4) flash attention (48*197 waves, 4 waves/block)
  {
    int blks = (48 * QTILES) / 4;   // 2364
    attn_kernel<<<blks, 128, 0, stream>>>(qbf0, kbf0, vt0, obf0);
    attn_kernel<<<blks, 128, 0, stream>>>(qbf1, kbf1, vt1, obf1);
  }

  // 5) output projection + shape
  float* xo  = (float*)d_out;
  float* shp = xo + (size_t)MROWS * DIMX;
  float* yo  = shp + 3;
  {
    int waves = (MPAD / 16) * (DIMX / 16);
    int blks = waves / 8;           // 4710
    gemm_proj_kernel<<<blks, 256, 0, stream>>>(obf0, pwb, proj_b, xo);
    gemm_proj_kernel<<<blks, 256, 0, stream>>>(obf1, pwb, proj_b, yo);
  }
  write_shape_kernel<<<1, 32, 0, stream>>>(shp);
}
